// SphereLoss_10333691314718
// MI455X (gfx1250) — matrix operand order repaired and checked by reading, not compiled, not run on hardware
//
#include <hip/hip_runtime.h>
#include <hip/hip_bf16.h>
#include <math.h>

#define BATCH 512
#define DIMK  512
#define NCLS  85742

typedef __bf16  bf16x16 __attribute__((ext_vector_type(16)));
typedef __bf16  bf16x2  __attribute__((ext_vector_type(2)));
typedef float   f32x8   __attribute__((ext_vector_type(8)));
typedef float   f32x2   __attribute__((ext_vector_type(2)));
typedef int     v4i     __attribute__((ext_vector_type(4)));

typedef __attribute__((address_space(1))) v4i as1_v4i;   // global int4
typedef __attribute__((address_space(3))) v4i as3_v4i;   // LDS int4

#if defined(__gfx1250__) && __has_builtin(__builtin_amdgcn_global_load_async_to_lds_b128)
#define HAVE_ASYNC_LDS 1
#else
#define HAVE_ASYNC_LDS 0
#endif

// ws layout (bytes)
#define XN_OFF    0ull            // 512*512 bf16        = 524288
#define FN_OFF    524288ull       // 512 f32
#define RS_OFF    526336ull       // 512 f32
#define OL_OFF    528384ull       // 512 f32
#define WINV_OFF  530432ull       // 85742 f32 (342968)
#define WBF_OFF   873472ull       // 85742*512 bf16 = 87799808
#define WS_NEED   (WBF_OFF + 87799808ull)

union FragU { uint4 u[2]; bf16x16 v; };

// packed hw convert: float2 -> bf16x2 (v_cvt_pk_bf16_f32)
__device__ __forceinline__ unsigned int pack2_bf16(float a, float b) {
  f32x2 f; f.x = a; f.y = b;
  union { bf16x2 h; unsigned int u; } cv;
  cv.h = __builtin_convertvector(f, bf16x2);
  return cv.u;
}

__device__ __forceinline__ void wait_async0() {
#if defined(__gfx1250__)
#if __has_builtin(__builtin_amdgcn_s_wait_asynccnt)
  __builtin_amdgcn_s_wait_asynccnt(0);
#else
  asm volatile("s_wait_asynccnt 0x0" ::: "memory");
#endif
#endif
}

__device__ __forceinline__ void async_b128(const void* g, unsigned ldsOff, int immOff) {
#if HAVE_ASYNC_LDS
  if (immOff == 0)
    __builtin_amdgcn_global_load_async_to_lds_b128(
        (as1_v4i*)(unsigned long long)g, (as3_v4i*)ldsOff, 0, 0);
  else
    __builtin_amdgcn_global_load_async_to_lds_b128(
        (as1_v4i*)(unsigned long long)g, (as3_v4i*)ldsOff, 16, 0);
#endif
}

// ---------------- Kernel 1: x row-normalize (bf16), fnorm, zero accumulators
__global__ void xnorm_k(const float* __restrict__ x, unsigned short* __restrict__ xn,
                        float* __restrict__ fnorm, float* __restrict__ rowsum,
                        float* __restrict__ outlab) {
  const int b = blockIdx.x, t = threadIdx.x;          // 256 threads
  float2 v = ((const float2*)(x + (size_t)b * DIMK))[t];
  float ss = v.x * v.x + v.y * v.y;
#pragma unroll
  for (int m = 16; m >= 1; m >>= 1) ss += __shfl_xor(ss, m, 32);
  __shared__ float red[8];
  __shared__ float bnorm;
  if ((t & 31) == 0) red[t >> 5] = ss;
  __syncthreads();
  if (t == 0) {
    float tot = 0.f;
#pragma unroll
    for (int i = 0; i < 8; ++i) tot += red[i];
    float nrm = sqrtf(tot);
    fnorm[b]  = nrm;
    rowsum[b] = 0.f;
    outlab[b] = 0.f;
    bnorm = nrm;
  }
  __syncthreads();
  float inv = 1.0f / fmaxf(bnorm, 1e-12f);
  ((unsigned int*)xn)[b * (DIMK / 2) + t] = pack2_bf16(v.x * inv, v.y * inv);
}

// ---------------- Kernel 2: inverse row norms (+ optional bf16 weight copy)
__global__ void wprep_k(const float* __restrict__ w, float* __restrict__ winv,
                        unsigned short* __restrict__ wbf /* may be null */) {
  int widx = blockIdx.x * 8 + (threadIdx.x >> 5);
  if (widx >= NCLS) return;
  int lane = threadIdx.x & 31;
  const float4* rp = (const float4*)(w + (size_t)widx * DIMK + lane * 16);
  float4 v[4];
#pragma unroll
  for (int i = 0; i < 4; ++i) v[i] = rp[i];           // 16 contiguous floats / lane
  float ss = 0.f;
#pragma unroll
  for (int i = 0; i < 4; ++i)
    ss += v[i].x * v[i].x + v[i].y * v[i].y + v[i].z * v[i].z + v[i].w * v[i].w;
#pragma unroll
  for (int m = 16; m >= 1; m >>= 1) ss += __shfl_xor(ss, m, 32);
  if (lane == 0) winv[widx] = 1.0f / fmaxf(sqrtf(ss), 1e-12f);
  if (wbf) {
    const float* f = (const float*)v;
    unsigned int pk[8];
#pragma unroll
    for (int i = 0; i < 8; ++i) pk[i] = pack2_bf16(f[2 * i], f[2 * i + 1]);
    uint4* dst = (uint4*)(wbf + (size_t)widx * DIMK + lane * 16);
    dst[0] = make_uint4(pk[0], pk[1], pk[2], pk[3]);
    dst[1] = make_uint4(pk[4], pk[5], pk[6], pk[7]);
  }
}

// ---------------- Kernel 3: fused bf16-WMMA GEMM + SphereFace margin + online softmax sums
// PRE=1: weights pre-converted to bf16, both tiles DMA'd async global->LDS.
// PRE=0: weights fp32, converted in-kernel via v_cvt_pk_bf16_f32.
template <int PRE>
__launch_bounds__(256)
__global__ void gemm_k(const unsigned short* __restrict__ xn,
                       const float* __restrict__ weight,
                       const unsigned short* __restrict__ wbf,
                       const float* __restrict__ winv,
                       const float* __restrict__ fnorm,
                       const int* __restrict__ label,
                       float* __restrict__ rowsum,
                       float* __restrict__ outlab) {
  constexpr int LSTR = 40;                          // padded bf16 row stride
  __shared__ alignas(16) unsigned short lA[2][128 * LSTR];
  __shared__ alignas(16) unsigned short lW[2][128 * LSTR];
  __shared__ float lRow[128];

  const int t = threadIdx.x;
  const int lane = t & 31, wave = t >> 5;
  const int wm = wave >> 1, wn = wave & 1;          // 4 waves in M, 2 in N
  const int mbase = blockIdx.y * 128;
  const int cbase = blockIdx.x * 128;
  const int lm16 = lane & 15, lh = lane >> 4;

  if (t < 128) lRow[t] = 0.f;

  // global->LDS loaders: each thread owns one 16-elem half-row of the 128x32 tile
  const int lrow = t >> 1, lcol = (t & 1) * 16;
  const int wrow = (NCLS - 1 < cbase + lrow) ? (NCLS - 1) : (cbase + lrow); // clamp: OOB cols masked in epilogue
  float4 wR[4];
  uint4  aR0, aR1, wB0, wB1;

  auto issueA = [&](int kt, int b2) {
    const unsigned short* gp = xn + (size_t)(mbase + lrow) * DIMK + kt * 32 + lcol;
#if HAVE_ASYNC_LDS
    unsigned lp = (unsigned)(unsigned long long)&lA[b2][lrow * LSTR + lcol];
    async_b128(gp, lp, 0);
    async_b128(gp, lp, 16);
#else
    aR0 = ((const uint4*)gp)[0]; aR1 = ((const uint4*)gp)[1];
    (void)b2;
#endif
  };
  auto issueW = [&](int kt, int b2) {
    if constexpr (PRE) {
      const unsigned short* gp = wbf + (size_t)wrow * DIMK + kt * 32 + lcol;
#if HAVE_ASYNC_LDS
      unsigned lp = (unsigned)(unsigned long long)&lW[b2][lrow * LSTR + lcol];
      async_b128(gp, lp, 0);
      async_b128(gp, lp, 16);
#else
      wB0 = ((const uint4*)gp)[0]; wB1 = ((const uint4*)gp)[1];
      (void)b2;
#endif
    } else {
      const float4* p = (const float4*)(weight + (size_t)wrow * DIMK + kt * 32 + lcol);
      wR[0] = p[0]; wR[1] = p[1]; wR[2] = p[2]; wR[3] = p[3];
      (void)b2;
    }
  };
  auto stage = [&](int b2) {                        // VGPR->LDS staging (non-async parts)
#if !HAVE_ASYNC_LDS
    uint4* da = (uint4*)&lA[b2][lrow * LSTR + lcol];
    da[0] = aR0; da[1] = aR1;
#endif
    if constexpr (PRE) {
#if !HAVE_ASYNC_LDS
      uint4* dw = (uint4*)&lW[b2][lrow * LSTR + lcol];
      dw[0] = wB0; dw[1] = wB1;
#endif
    } else {
      const float* wf = (const float*)wR;
      unsigned int pk[8];
#pragma unroll
      for (int i = 0; i < 8; ++i) pk[i] = pack2_bf16(wf[2 * i], wf[2 * i + 1]);
      uint4* dw = (uint4*)&lW[b2][lrow * LSTR + lcol];
      dw[0] = make_uint4(pk[0], pk[1], pk[2], pk[3]);
      dw[1] = make_uint4(pk[4], pk[5], pk[6], pk[7]);
    }
  };

  f32x8 acc[2][4];
#pragma unroll
  for (int i = 0; i < 2; ++i)
#pragma unroll
    for (int j = 0; j < 4; ++j) acc[i][j] = (f32x8)(0.0f);

  issueA(0, 0);
  issueW(0, 0);
  int buf = 0;
  for (int kt = 0; kt < DIMK / 32; ++kt) {
    stage(buf);
    wait_async0();                                   // my wave's async tiles landed
    __syncthreads();                                 // everyone's A+W tile visible
    if (kt < DIMK / 32 - 1) {
      issueA(kt + 1, buf ^ 1);
      issueW(kt + 1, buf ^ 1);
    }

    FragU aF[2], bF[4];
#pragma unroll
    for (int i = 0; i < 2; ++i) {                    // ISA 16-bit A 16x32 layout
      const unsigned short* p = &lA[buf][(wm * 32 + i * 16 + lm16) * LSTR];
      aF[i].u[0] = *(const uint4*)(p + lh * 8);      // K = lh*8 .. +7
      aF[i].u[1] = *(const uint4*)(p + 16 + lh * 8); // K = 16+lh*8 .. +7
    }
#pragma unroll
    for (int j = 0; j < 4; ++j) {                    // ISA 16-bit B 32x16 layout
      const unsigned short* p = &lW[buf][(wn * 64 + j * 16 + lm16) * LSTR + lh * 8];
      bF[j].u[0] = *(const uint4*)(p);               // K = lh*8 .. +7
      bF[j].u[1] = *(const uint4*)(p + 16);          // K = 16+lh*8 .. +7
    }
#pragma unroll
    for (int i = 0; i < 2; ++i)
#pragma unroll
      for (int j = 0; j < 4; ++j)
        acc[i][j] = __builtin_amdgcn_wmma_f32_16x16x32_bf16(
            false, aF[i].v, false, bF[j].v, (short)0, acc[i][j], false, false);
    buf ^= 1;
  }

  // -------- epilogue: cos -> (margin at label) -> exp(out - fnorm) partial sums
  float wv[4];
#pragma unroll
  for (int j = 0; j < 4; ++j) {
    int gc = cbase + wn * 64 + j * 16 + lm16;
    wv[j] = (gc < NCLS) ? winv[gc] : 0.f;
  }
  const float INV1PL = 1.0f / 893.8571428571429f;    // 1/(1+lamb), lamb = 1000/1.12
#pragma unroll
  for (int i = 0; i < 2; ++i) {
#pragma unroll
    for (int r = 0; r < 8; ++r) {
      int lm = wm * 32 + i * 16 + r + 8 * lh;        // C/D layout: row = r + 8*(lane>>4)
      int gm = mbase + lm;
      float fn = fnorm[gm];
      int lb = label[gm];
      float part = 0.f;
#pragma unroll
      for (int j = 0; j < 4; ++j) {
        int gc = cbase + wn * 64 + j * 16 + lm16;
        if (gc < NCLS) {
          float cosv = fminf(fmaxf(acc[i][j][r] * wv[j], -1.f), 1.f);
          float o = cosv * fn;
          if (gc == lb) {
            float c2 = cosv * cosv;
            float cm = 8.f * c2 * c2 - 8.f * c2 + 1.f;          // cos(4*theta)
            float kk = floorf(4.0f * acosf(cosv) * (1.0f / 3.14159265f));
            float sgn = (((int)kk) & 1) ? -1.f : 1.f;
            float phi = sgn * cm - 2.f * kk;
            o = ((phi - cosv) * INV1PL + cosv) * fn;
            outlab[gm] = o;                            // unique owner: plain store
          }
          part += __expf(o - fn);                      // fn is a valid stabilizer
        }
      }
      part += __shfl_xor(part, 1, 32);
      part += __shfl_xor(part, 2, 32);
      part += __shfl_xor(part, 4, 32);
      part += __shfl_xor(part, 8, 32);
      if (lm16 == 0) atomicAdd(&lRow[lm], part);
    }
  }
  __syncthreads();
  if (t < 128) atomicAdd(&rowsum[mbase + t], lRow[t]);
}

// ---------------- Kernel 4: loss = mean(fnorm + log(rowsum) - out_label)
__global__ void loss_k(const float* __restrict__ fnorm, const float* __restrict__ rowsum,
                       const float* __restrict__ outlab, float* __restrict__ out) {
  int t = threadIdx.x;                               // 512 threads
  float v = fnorm[t] + logf(rowsum[t]) - outlab[t];
#pragma unroll
  for (int m = 16; m >= 1; m >>= 1) v += __shfl_xor(v, m, 32);
  __shared__ float red[16];
  if ((t & 31) == 0) red[t >> 5] = v;
  __syncthreads();
  if (t == 0) {
    float tot = 0.f;
#pragma unroll
    for (int i = 0; i < 16; ++i) tot += red[i];
    out[0] = tot / (float)BATCH;
  }
}

extern "C" void kernel_launch(void* const* d_in, const int* in_sizes, int n_in,
                              void* d_out, int out_size, void* d_ws, size_t ws_size,
                              hipStream_t stream) {
  (void)in_sizes; (void)n_in; (void)out_size;
  const float* x      = (const float*)d_in[0];
  const float* weight = (const float*)d_in[1];
  const int*   label  = (const int*)d_in[2];
  char* ws = (char*)d_ws;
  unsigned short* xn = (unsigned short*)(ws + XN_OFF);
  float* fnorm  = (float*)(ws + FN_OFF);
  float* rowsum = (float*)(ws + RS_OFF);
  float* outlab = (float*)(ws + OL_OFF);
  float* winv   = (float*)(ws + WINV_OFF);
  unsigned short* wbf = (ws_size >= WS_NEED) ? (unsigned short*)(ws + WBF_OFF) : nullptr;
  float* out = (float*)d_out;

  xnorm_k<<<BATCH, 256, 0, stream>>>(x, xn, fnorm, rowsum, outlab);
  wprep_k<<<(NCLS + 7) / 8, 256, 0, stream>>>(weight, winv, wbf);
  dim3 grid((NCLS + 127) / 128, BATCH / 128);
  if (wbf)
    gemm_k<1><<<grid, 256, 0, stream>>>(xn, weight, wbf, winv, fnorm, label, rowsum, outlab);
  else
    gemm_k<0><<<grid, 256, 0, stream>>>(xn, weight, nullptr, winv, fnorm, label, rowsum, outlab);
  loss_k<<<1, BATCH, 0, stream>>>(fnorm, rowsum, outlab, out);
}
